// TopGNN_53575422050967
// MI455X (gfx1250) — matrix operand order repaired and checked by reading, not compiled, not run on hardware
//
#include <hip/hip_runtime.h>
#include <hip/hip_bf16.h>

// Problem constants (match reference)
#define BB 8
#define HH 12
#define SS 512
#define DD 768
#define DP 768
#define KK 51

typedef __attribute__((ext_vector_type(16))) _Float16 v16h;
typedef __attribute__((ext_vector_type(8)))  float    v8f;
typedef __attribute__((ext_vector_type(4)))  float    v4f;

// ---------- helpers ----------
__device__ __forceinline__ unsigned sortable_u32(float f) {
    unsigned u = __float_as_uint(f);
    return (u & 0x80000000u) ? ~u : (u | 0x80000000u);
}

__device__ __forceinline__ unsigned long long wave_max_u64(unsigned long long x) {
#pragma unroll
    for (int off = 16; off >= 1; off >>= 1) {
        unsigned lo = (unsigned)x, hi = (unsigned)(x >> 32);
        unsigned lo2 = __shfl_xor(lo, off, 32);
        unsigned hi2 = __shfl_xor(hi, off, 32);
        unsigned long long y = (((unsigned long long)hi2) << 32) | lo2;
        x = (x > y) ? x : y;
    }
    return x;
}

// ---------- kernel Z: zero coeff accumulator ----------
__global__ void zero_ws(float* p, int n) {
    int i = blockIdx.x * blockDim.x + threadIdx.x;
    if (i < n) p[i] = 0.0f;
}

// ---------- kernel A: head-mean + top-K + mask -> scatter scalar coefficients ----------
// one wave32 per (branch, b, s) row. coeff layout [2][B][S]
// lane-blocked layout: column j = lane*16 + r  ->  4x b128 NT loads per head per lane
__global__ void topk_scatter(const float* __restrict__ attW, const float* __restrict__ attS,
                             const int* __restrict__ lens,
                             const float* __restrict__ etaW, const float* __restrict__ etaS,
                             float* __restrict__ coeff) {
    const int lane = threadIdx.x & 31;
    const int wid  = blockIdx.x * (blockDim.x >> 5) + (threadIdx.x >> 5);
    const int branch = wid / (BB * SS);
    const int rem    = wid - branch * (BB * SS);
    const int b = rem >> 9;      // / 512
    const int s = rem & 511;
    const int len = lens[b];
    if (s >= len) return;        // tgt_valid==0 -> contributes nothing (uniform per wave)

    const float* att = branch ? attS : attW;
    const float  eta = branch ? etaS[0] : etaW[0];
    const float* rowbase = att + (size_t)b * (HH * SS * SS) + (size_t)s * SS;

    // sum over 12 heads (sign/ordering identical to mean); stream with NT b128 loads
    v4f acc[4];
#pragma unroll
    for (int q = 0; q < 4; ++q) acc[q] = (v4f){0.f, 0.f, 0.f, 0.f};
    for (int h = 0; h < HH; ++h) {
        const v4f* p4 = reinterpret_cast<const v4f*>(rowbase + (size_t)h * (SS * SS)) + lane * 4;
        if (h + 1 < HH)
            __builtin_prefetch(rowbase + (size_t)(h + 1) * (SS * SS) + lane * 16, 0, 0);
#pragma unroll
        for (int q = 0; q < 4; ++q) acc[q] += __builtin_nontemporal_load(p4 + q);
    }

    // packed keys: (sortable(value) << 32) | (j ^ 0x1FF); max => largest value, ties -> smallest j
    unsigned long long pks[16];
#pragma unroll
    for (int q = 0; q < 4; ++q) {
#pragma unroll
        for (int i = 0; i < 4; ++i) {
            const int r = 4 * q + i;
            const int j = lane * 16 + r;
            pks[r] = (((unsigned long long)sortable_u32(acc[q][i])) << 32) |
                     (unsigned)(j ^ 0x1FF);
        }
    }

    int mysel0 = -1, mysel1 = -1;
    int cnt = 0;
    bool anyPos = false;
    for (int k = 0; k < KK; ++k) {
        unsigned long long best = 0ULL;
#pragma unroll
        for (int r = 0; r < 16; ++r)
            if (pks[r] > best) best = pks[r];
        best = wave_max_u64(best);
        const unsigned keymax = (unsigned)(best >> 32);
        const int j = (((int)best) & 0x1FF) ^ 0x1FF;
        const bool pos = keymax > 0x80000000u;   // strictly > 0.0f
        if (k == 0) anyPos = pos;                // vals sorted desc => any positive iff first is
        const bool keep = (anyPos ? pos : true) && (j < len);
        if (keep) ++cnt;
        if ((k & 31) == lane) {
            if ((k >> 5) == 0) mysel0 = keep ? j : -1;
            else               mysel1 = keep ? j : -1;
        }
        if ((j >> 4) == lane) {                  // owner lane removes winner
            const int rr = j & 15;
#pragma unroll
            for (int r = 0; r < 16; ++r)
                if (r == rr) pks[r] = 0ULL;
        }
    }

    const float denom = (float)(cnt > 0 ? cnt : 1);
    const float scale = (1.0f - eta) / ((float)len * denom);
    float* crow = coeff + (size_t)(branch * BB + b) * SS;
    if (mysel0 >= 0) atomicAdd(crow + mysel0, scale);
    if (mysel1 >= 0) atomicAdd(crow + mysel1, scale);
    if (lane == 0)   atomicAdd(crow + s, eta / (float)len);
}

// ---------- kernel B: pooledHidden[br][row][d] = sum_j coeff[br][b][j]*hidden[b][j][d]; csum ----------
// grid 2*16 blocks (rows 8..15 are zero padding for 16-row WMMA tiles), blockDim 256
__global__ void pool_hidden(const float* __restrict__ wEmb, const float* __restrict__ sEmb,
                            const float* __restrict__ coeff,
                            float* __restrict__ PH, float* __restrict__ csum) {
    const int branch = blockIdx.x >> 4;
    const int b = blockIdx.x & 15;
    const int tid = threadIdx.x;
    float* phRow = PH + (size_t)(branch * 16 + b) * DD;
    if (b >= 8) {
        phRow[tid] = 0.0f; phRow[tid + 256] = 0.0f; phRow[tid + 512] = 0.0f;
        if (tid == 0) csum[branch * 16 + b] = 0.0f;
        return;
    }
    const float* hid  = branch ? sEmb : wEmb;
    const float* crow = coeff + (size_t)(branch * BB + b) * SS;
    __shared__ float cs[SS];
    cs[tid] = crow[tid];
    cs[tid + 256] = crow[tid + 256];
    __syncthreads();
    float a0 = 0.f, a1 = 0.f, a2 = 0.f;
    for (int j = 0; j < SS; ++j) {
        const float* hr = hid + (size_t)(b * SS + j) * DD;
        const float c = cs[j];
        a0 += c * hr[tid];
        a1 += c * hr[tid + 256];
        a2 += c * hr[tid + 512];
    }
    phRow[tid] = a0; phRow[tid + 256] = a1; phRow[tid + 512] = a2;
    if (tid == 0) {
        float t = 0.f;
        for (int j = 0; j < SS; ++j) t += cs[j];
        csum[branch * 16 + b] = t;
    }
}

// ---------- shared WMMA tile: out16x16 = A[16][768] * W[ntile*16..+16][768]^T ----------
__device__ __forceinline__ void gemm_tile16(const float* __restrict__ A, const float* __restrict__ W,
                                            int ntile, int lane, float out[8]) {
    const int mA = lane & 15;
    const int hs = lane >> 4;
    const float* arow = A + (size_t)mA * 768;
    const float* wrow = W + (size_t)(ntile * 16 + (lane & 15)) * 768;
    v8f c = {0.f, 0.f, 0.f, 0.f, 0.f, 0.f, 0.f, 0.f};
    for (int kb = 0; kb < 768; kb += 64) {
        v16h a0, b0, a1, b1;
#pragma unroll
        for (int v = 0; v < 8; ++v) {
            const int ka = ((v < 4) ? (2 * v) : (2 * v + 8)) + 8 * hs;  // A 16x32 f16 layout
            const int kn = ((v < 4) ? (2 * v) : (2 * v + 8)) + 8 * hs;  // B 32x16 f16 layout (same k order as A)
            a0[2 * v]     = (_Float16)arow[kb + ka];
            a0[2 * v + 1] = (_Float16)arow[kb + ka + 1];
            b0[2 * v]     = (_Float16)wrow[kb + kn];
            b0[2 * v + 1] = (_Float16)wrow[kb + kn + 1];
            a1[2 * v]     = (_Float16)arow[kb + 32 + ka];
            a1[2 * v + 1] = (_Float16)arow[kb + 32 + ka + 1];
            b1[2 * v]     = (_Float16)wrow[kb + 32 + kn];
            b1[2 * v + 1] = (_Float16)wrow[kb + 32 + kn + 1];
        }
        c = __builtin_amdgcn_wmma_f32_16x16x32_f16(false, a0, false, b0, (short)0, c, false, false);
        c = __builtin_amdgcn_wmma_f32_16x16x32_f16(false, a1, false, b1, (short)0, c, false, false);
    }
#pragma unroll
    for (int r = 0; r < 8; ++r) out[r] = c[r];
}

// ---------- C1: t1 = tanh(PH @ Wt^T + bt*csum) ----------
__global__ void gemm_trans_tanh(const float* __restrict__ PH,
                                const float* __restrict__ W0, const float* __restrict__ W1,
                                const float* __restrict__ bt0, const float* __restrict__ bt1,
                                const float* __restrict__ csum, float* __restrict__ t1) {
    const int branch = blockIdx.x / 48;
    const int ntile  = blockIdx.x % 48;
    const int lane   = threadIdx.x;
    const float* A  = PH + (size_t)branch * 16 * 768;
    const float* W  = branch ? W1 : W0;
    const float* bt = branch ? bt1 : bt0;
    float o[8];
    gemm_tile16(A, W, ntile, lane, o);
    const int n = lane & 15, mb = (lane >> 4) * 8, p = ntile * 16 + n;
    const float btp = bt[p];
#pragma unroll
    for (int r = 0; r < 8; ++r) {
        const int m = mb + r;
        const float val = o[r] + btp * csum[branch * 16 + m];
        t1[(size_t)(branch * 16 + m) * 768 + p] = tanhf(val);
    }
}

// ---------- C2: fco = t1 @ Wf^T + bf ----------
__global__ void gemm_fc(const float* __restrict__ t1,
                        const float* __restrict__ W0, const float* __restrict__ W1,
                        const float* __restrict__ bf0, const float* __restrict__ bf1,
                        float* __restrict__ fco) {
    const int branch = blockIdx.x / 48;
    const int ntile  = blockIdx.x % 48;
    const int lane   = threadIdx.x;
    const float* A  = t1 + (size_t)branch * 16 * 768;
    const float* W  = branch ? W1 : W0;
    const float* bf = branch ? bf1 : bf0;
    float o[8];
    gemm_tile16(A, W, ntile, lane, o);
    const int n = lane & 15, mb = (lane >> 4) * 8, p = ntile * 16 + n;
    const float bp = bf[p];
#pragma unroll
    for (int r = 0; r < 8; ++r) {
        const int m = mb + r;
        fco[(size_t)(branch * 16 + m) * 768 + p] = o[r] + bp;
    }
}

// ---------- C3: LayerNorm rows (only the 8 real rows per branch) ----------
__global__ void ln_rows(const float* __restrict__ fco,
                        const float* __restrict__ g0, const float* __restrict__ b0,
                        const float* __restrict__ g1, const float* __restrict__ b1,
                        float* __restrict__ out) {
    const int branch = blockIdx.x >> 3;
    const int b = blockIdx.x & 7;
    const int tid = threadIdx.x;
    const float* x = fco + (size_t)(branch * 16 + b) * 768;
    float*       y = out + (size_t)(branch * 16 + b) * 768;
    const float* g  = branch ? g1 : g0;
    const float* bb = branch ? b1 : b0;
    const float x0 = x[tid], x1 = x[tid + 256], x2 = x[tid + 512];
    __shared__ float red[256];
    red[tid] = x0 + x1 + x2;
    __syncthreads();
    for (int off = 128; off; off >>= 1) { if (tid < off) red[tid] += red[tid + off]; __syncthreads(); }
    const float mu = red[0] * (1.0f / 768.0f);
    __syncthreads();
    const float d0 = x0 - mu, d1 = x1 - mu, d2 = x2 - mu;
    red[tid] = d0 * d0 + d1 * d1 + d2 * d2;
    __syncthreads();
    for (int off = 128; off; off >>= 1) { if (tid < off) red[tid] += red[tid + off]; __syncthreads(); }
    const float var  = red[0] * (1.0f / 768.0f);
    const float rstd = rsqrtf(var + 1e-5f);
    y[tid]       = d0 * rstd * g[tid]       + bb[tid];
    y[tid + 256] = d1 * rstd * g[tid + 256] + bb[tid + 256];
    y[tid + 512] = d2 * rstd * g[tid + 512] + bb[tid + 512];
}

// ---------- C4: final = tanh(ln @ fc2^T + b2) -> d_out ----------
__global__ void gemm_final(const float* __restrict__ ln,
                           const float* __restrict__ W0, const float* __restrict__ W1,
                           const float* __restrict__ b20, const float* __restrict__ b21,
                           float* __restrict__ outp) {
    const int branch = blockIdx.x / 48;
    const int ntile  = blockIdx.x % 48;
    const int lane   = threadIdx.x;
    const float* A  = ln + (size_t)branch * 16 * 768;
    const float* W  = branch ? W1 : W0;
    const float* b2 = branch ? b21 : b20;
    float o[8];
    gemm_tile16(A, W, ntile, lane, o);
    const int n = lane & 15, mb = (lane >> 4) * 8, p = ntile * 16 + n;
    const float bp = b2[p];
#pragma unroll
    for (int r = 0; r < 8; ++r) {
        const int m = mb + r;
        if (m < 8)
            outp[((size_t)branch * 8 + m) * 768 + p] = tanhf(o[r] + bp);
    }
}

extern "C" void kernel_launch(void* const* d_in, const int* in_sizes, int n_in,
                              void* d_out, int out_size, void* d_ws, size_t ws_size,
                              hipStream_t stream) {
    const float* wAtt = (const float*)d_in[0];
    const float* wEmb = (const float*)d_in[1];
    const float* sAtt = (const float*)d_in[2];
    const float* sEmb = (const float*)d_in[3];
    const int*   lens = (const int*)d_in[4];
    const float* wTw  = (const float*)d_in[5];
    const float* wTb  = (const float*)d_in[6];
    const float* wFw  = (const float*)d_in[7];
    const float* wFb  = (const float*)d_in[8];
    const float* wLg  = (const float*)d_in[9];
    const float* wLb  = (const float*)d_in[10];
    const float* wEta = (const float*)d_in[11];
    const float* sTw  = (const float*)d_in[12];
    const float* sTb  = (const float*)d_in[13];
    const float* sFw  = (const float*)d_in[14];
    const float* sFb  = (const float*)d_in[15];
    const float* sLg  = (const float*)d_in[16];
    const float* sLb  = (const float*)d_in[17];
    const float* sEta = (const float*)d_in[18];
    const float* fcWw = (const float*)d_in[19];
    const float* fcWb = (const float*)d_in[20];
    const float* fcSw = (const float*)d_in[21];
    const float* fcSb = (const float*)d_in[22];

    float* ws    = (float*)d_ws;
    float* coeff = ws;                      // [2][B][S] = 8192
    float* csum  = ws + 8192;               // [2][16]   = 32
    float* PH    = ws + 8224;               // [2][16][768]
    float* t1    = PH + 2 * 16 * 768;
    float* fco   = t1 + 2 * 16 * 768;
    float* lnb   = fco + 2 * 16 * 768;

    zero_ws<<<(2 * BB * SS + 255) / 256, 256, 0, stream>>>(coeff, 2 * BB * SS);
    topk_scatter<<<(2 * BB * SS) / 8, 256, 0, stream>>>(wAtt, sAtt, lens, wEta, sEta, coeff);
    pool_hidden<<<32, 256, 0, stream>>>(wEmb, sEmb, coeff, PH, csum);
    gemm_trans_tanh<<<96, 32, 0, stream>>>(PH, wTw, sTw, wTb, sTb, csum, t1);
    gemm_fc<<<96, 32, 0, stream>>>(t1, wFw, sFw, wFb, sFb, fco);
    ln_rows<<<16, 256, 0, stream>>>(fco, wLg, wLb, sLg, sLb, lnb);
    gemm_final<<<96, 32, 0, stream>>>(lnb, fcWw, fcSw, fcWb, fcSb, (float*)d_out);
}
